// SeqCondAttention_79783312491004
// MI455X (gfx1250) — matrix operand order repaired and checked by reading, not compiled, not run on hardware
//
#include <hip/hip_runtime.h>
#include <hip/hip_bf16.h>

// ---------------------------------------------------------------------------
// Model constants: B=4, L=4096, D=768, K=12, K_Q=6, H=64, KSZ=4
// DIM_TOTAL=1752, DIM_CONV_TOTAL=1548, oc_flat=1536, y per token=2304
// ---------------------------------------------------------------------------

typedef __attribute__((ext_vector_type(16))) __bf16 v16bf;
typedef __attribute__((ext_vector_type(8)))  __bf16 v8bf;
typedef __attribute__((ext_vector_type(8)))  float  v8f;
typedef __attribute__((ext_vector_type(4)))  unsigned int v4u;
typedef __attribute__((ext_vector_type(8)))  int v8i;
typedef __attribute__((ext_vector_type(4)))  int v4i;

__device__ __forceinline__ float sigmoidf_(float x) { return 1.0f / (1.0f + __expf(-x)); }
__device__ __forceinline__ float softplusf_(float x) { return (x > 20.0f) ? x : log1pf(__expf(x)); }

// ---------------------------------------------------------------------------
// TDM: DMA a 2D bf16 tile (tile_dim0=32 elems per row, tile_dim1=64 rows) from
// global into LDS.  Hardware pads 16B after every 64B row -> LDS row stride of
// 40 bf16.  Rows/cols beyond (tdim1, tdim0) are zero-filled (edge tiles).
// D# encoding per CDNA5 ISA §8.3/§8.4; 6-arg builtin (groups 0-3 + cpol),
// groups 2/3 zero for 2D tensors.
// ---------------------------------------------------------------------------
__device__ __forceinline__ void tdm_load_tile_64x32_bf16(
    unsigned lds_off, const __bf16* gptr, unsigned tdim0, unsigned tdim1,
    unsigned long long stride_elems)
{
    unsigned long long ga = (unsigned long long)(uintptr_t)gptr;
    v4u g0;
    g0[0] = 1u;                                            // count=1 (valid D#)
    g0[1] = lds_off;                                       // lds_addr (bytes)
    g0[2] = (unsigned)(ga & 0xFFFFFFFFull);                // global_addr[31:0]
    g0[3] = (unsigned)((ga >> 32) & 0x01FFFFFFull)         // global_addr[56:32]
          | 0x80000000u;                                   // type=2 ("image")
    v8i g1;
    g1[0] = (int)((1u << 16)      // data_size = 1 -> 2 bytes
                | (1u << 20)      // pad_enable
                | (3u << 22)      // pad_interval = 16 DWORDs (64B)
                | (3u << 25));    // pad_amount   = 4 DWORDs (16B)
    g1[1] = (int)((tdim0 & 0xFFFFu) << 16);                // tensor_dim0[15:0]
    g1[2] = (int)((tdim0 >> 16) | ((tdim1 & 0xFFFFu) << 16));
    g1[3] = (int)((tdim1 >> 16) | (32u << 16));            // tile_dim0 = 32
    g1[4] = (int)64u;                                      // tile_dim1 = 64
    g1[5] = (int)(unsigned)(stride_elems & 0xFFFFFFFFull); // dim0 stride lo
    g1[6] = (int)(unsigned)((stride_elems >> 32) & 0xFFFFull);
    g1[7] = 0;
    v4i z4 = {};
    v8i z8 = {};
    __builtin_amdgcn_tensor_load_to_lds(g0, g1, z4, z4, z8, 0);
}

// ---------------------------------------------------------------------------
// Strided-batched GEMM, C[M,N](f32) = A[M,K](bf16) * Bt[N,K](bf16)^T with
// v_wmma_f32_16x16x32_bf16 and double-buffered TDM pipelining:
// iteration i waits (tensorcnt) for buffer i&1, then immediately issues the
// DMA pair for buffer (i+1)&1, overlapping the TDM with the WMMA work.
// Requires K % 32 == 0.  Block: 128 threads (4 waves), 64x64 tile.
// ---------------------------------------------------------------------------
__global__ __launch_bounds__(128)
void gemm_bf16_wmma_tdm(const __bf16* __restrict__ A, const __bf16* __restrict__ Bt,
                        float* __restrict__ C, int M, int N, int K,
                        int lda, int ldbt, int ldc,
                        long sA, long sB, long sC, int accumulate)
{
    __shared__ __align__(16) __bf16 As[2][64][40];  // TDM-padded rows (80B)
    __shared__ __align__(16) __bf16 Bs[2][64][40];

    const int tid  = threadIdx.x;
    const int wave = tid >> 5;
    const int lane = tid & 31;
    const int lr   = lane & 15;
    const int half = lane >> 4;

    const int m0 = blockIdx.y * 64;
    const int n0 = blockIdx.x * 64;
    A  += (size_t)blockIdx.z * sA;
    Bt += (size_t)blockIdx.z * sB;
    C  += (size_t)blockIdx.z * sC;

    const int wm = (wave >> 1) * 32;
    const int wn = (wave & 1) * 32;

    v8f acc[2][2] = {};

    // prologue: prefetch first tile pair into buffer 0
    if (tid < 32) {
        tdm_load_tile_64x32_bf16(
            (unsigned)(uintptr_t)&As[0][0][0], A + (size_t)m0 * lda,
            (unsigned)K, (unsigned)(M - m0), (unsigned long long)lda);
        tdm_load_tile_64x32_bf16(
            (unsigned)(uintptr_t)&Bs[0][0][0], Bt + (size_t)n0 * ldbt,
            (unsigned)K, (unsigned)(N - n0), (unsigned long long)ldbt);
    }

    int cur = 0;
    for (int kk = 0; kk < K; kk += 32, cur ^= 1) {
        if (tid < 32)
            __builtin_amdgcn_s_wait_tensorcnt(0);   // buffer `cur` landed
        __syncthreads();   // publishes buf[cur]; proves buf[cur^1] compute done

        // prefetch next tile pair into the other buffer (overlaps with WMMA)
        if (kk + 32 < K && tid < 32) {
            tdm_load_tile_64x32_bf16(
                (unsigned)(uintptr_t)&As[cur ^ 1][0][0],
                A + (size_t)m0 * lda + kk + 32,
                (unsigned)(K - kk - 32), (unsigned)(M - m0),
                (unsigned long long)lda);
            tdm_load_tile_64x32_bf16(
                (unsigned)(uintptr_t)&Bs[cur ^ 1][0][0],
                Bt + (size_t)n0 * ldbt + kk + 32,
                (unsigned)(K - kk - 32), (unsigned)(N - n0),
                (unsigned long long)ldbt);
        }

        // Fragments (ISA §7.12.2): A 16x32 per-lane K-groups {0-7,16-23}/{8-15,24-31};
        // B 32x16 per-lane column with the same K-groups {0-7,16-23}/{8-15,24-31}.
        v16bf afr[2], bfr[2];
        #pragma unroll
        for (int mi = 0; mi < 2; ++mi) {
            int row = wm + mi * 16 + lr;
            v8bf lo = *reinterpret_cast<const v8bf*>(&As[cur][row][half * 8]);
            v8bf hi = *reinterpret_cast<const v8bf*>(&As[cur][row][16 + half * 8]);
            #pragma unroll
            for (int j = 0; j < 8; ++j) { afr[mi][j] = lo[j]; afr[mi][8 + j] = hi[j]; }
        }
        #pragma unroll
        for (int ni = 0; ni < 2; ++ni) {
            int col = wn + ni * 16 + lr;
            v8bf lo = *reinterpret_cast<const v8bf*>(&Bs[cur][col][half * 8]);
            v8bf hi = *reinterpret_cast<const v8bf*>(&Bs[cur][col][16 + half * 8]);
            #pragma unroll
            for (int j = 0; j < 8; ++j) { bfr[ni][j] = lo[j]; bfr[ni][8 + j] = hi[j]; }
        }
        #pragma unroll
        for (int mi = 0; mi < 2; ++mi)
            #pragma unroll
            for (int ni = 0; ni < 2; ++ni)
                acc[mi][ni] = __builtin_amdgcn_wmma_f32_16x16x32_bf16(
                    false, afr[mi], false, bfr[ni], (short)0, acc[mi][ni], false, false);
    }

    #pragma unroll
    for (int mi = 0; mi < 2; ++mi)
        #pragma unroll
        for (int ni = 0; ni < 2; ++ni)
            #pragma unroll
            for (int r = 0; r < 8; ++r) {
                int row = m0 + wm + mi * 16 + half * 8 + r;
                int col = n0 + wn + ni * 16 + lr;
                if (row < M && col < N) {
                    size_t off = (size_t)row * ldc + col;
                    float v = acc[mi][ni][r];
                    C[off] = accumulate ? (C[off] + v) : v;
                }
            }
}

// ---------------------------------------------------------------------------
// f32 -> bf16 conversion / transpose helpers (one-shot, off the critical loop)
// ---------------------------------------------------------------------------
__global__ void cvt_bf16_kernel(const float* __restrict__ s, __bf16* __restrict__ d, long n)
{
    long i = (long)blockIdx.x * blockDim.x + threadIdx.x;
    if (i < n) d[i] = (__bf16)s[i];
}

// dst[b][n*K + k] = (bf16) src[b][k*N + n]
__global__ void transpose_cvt_kernel(const float* __restrict__ src, __bf16* __restrict__ dst,
                                     int K, int N, long sSrc, long sDst)
{
    long i = (long)blockIdx.x * blockDim.x + threadIdx.x;
    long total = (long)K * N;
    if (i >= total) return;
    int n = (int)(i / K);
    int k = (int)(i % K);
    dst[(size_t)blockIdx.y * sDst + i] =
        (__bf16)src[(size_t)blockIdx.y * sSrc + (size_t)k * N + n];
}

// compact c_skip slice of z_all into bf16 (BL x 192)
__global__ void gather_cskip_kernel(const float* __restrict__ zall, __bf16* __restrict__ d, long n)
{
    long i = (long)blockIdx.x * blockDim.x + threadIdx.x;
    if (i >= n) return;
    long bl = i / 192;
    int  j  = (int)(i % 192);
    d[i] = (__bf16)zall[bl * 1752 + 1548 + j];
}

// ---------------------------------------------------------------------------
// Fused depthwise causal conv(4) + SiLU + k/q RMSNorms + p_w + re/im stack.
// One workgroup (256 threads) per token; LDS reductions via ds_add_f32.
// ---------------------------------------------------------------------------
__global__ __launch_bounds__(256)
void conv_stack_kernel(const float* __restrict__ zall, const float* __restrict__ conv_w,
                       const float* __restrict__ k_norm_w, const float* __restrict__ q_norm_w,
                       const float* __restrict__ theta_raw, const float* __restrict__ decay_slopes,
                       const float* __restrict__ score_scale, const float* __restrict__ score_bias,
                       const float* __restrict__ tanh_scale,
                       float* __restrict__ stack, float* __restrict__ qn, int L)
{
    __shared__ float zc[1548];
    __shared__ float red[13];
    __shared__ float pw[12];

    const int  tid   = threadIdx.x;
    const long bl    = blockIdx.x;
    const int  l     = (int)(bl % L);
    const long zbase = bl * 1752;

    if (tid < 13) red[tid] = 0.0f;
    __syncthreads();

    for (int c = tid; c < 1548; c += 256) {
        float acc = 0.0f;
        #pragma unroll
        for (int j = 0; j < 4; ++j) {
            int ll = l - 3 + j;
            if (ll >= 0)
                acc += conv_w[j * 1548 + c] * zall[zbase + (long)(ll - l) * 1752 + c];
        }
        acc = acc * sigmoidf_(acc);          // SiLU
        zc[c] = acc;
        if (c < 768)       atomicAdd(&red[c >> 6], acc * acc);
        else if (c >= 780) atomicAdd(&red[12], acc * acc);
    }
    __syncthreads();

    float rq = rsqrtf(red[12] * (1.0f / 768.0f) + 1e-6f);
    for (int t = tid; t < 768; t += 256)
        qn[bl * 768 + t] = zc[780 + t] * rq * q_norm_w[t];

    if (tid < 12) {
        float sraw  = score_scale[tid] * zc[768 + tid] + score_bias[tid];
        float sp1   = softplusf_(sraw);
        float slope = softplusf_(decay_slopes[tid]);
        float dist  = (float)(L - 1 - l);
        float sp2   = softplusf_(-slope * dist);
        float p     = fminf(fmaxf(sp1 * sp2, 1e-6f), 1000.0f);
        pw[tid] = p;
        stack[bl * 1548 + tid] = p;
    }
    __syncthreads();

    for (int t = tid; t < 768; t += 256) {
        int   k   = t >> 6, h = t & 63;
        float kv  = zc[t] * rsqrtf(red[k] * (1.0f / 64.0f) + 1e-6f) * k_norm_w[h];
        float th  = 0.001f + 2.999f * sigmoidf_(theta_raw[t]);
        float phi = tanhf(kv * tanh_scale[k]) * th;
        float kvw = kv * pw[k];
        stack[bl * 1548 + 12  + t] = kvw * cosf(phi);
        stack[bl * 1548 + 780 + t] = kvw * sinf(phi);
    }
}

// ---------------------------------------------------------------------------
// In-place cumulative sum along L: one thread per (batch, channel).
// ---------------------------------------------------------------------------
__global__ void cumsum_kernel(float* __restrict__ s, int L)
{
    int c = blockIdx.x * blockDim.x + threadIdx.x;
    if (c >= 1548) return;
    size_t base = (size_t)blockIdx.y * L * 1548 + c;
    float acc = 0.0f;
    for (int l = 0; l < L; ++l) {
        size_t off = base + (size_t)l * 1548;
        acc += s[off];
        s[off] = acc;
    }
}

// ---------------------------------------------------------------------------
// Fused state normalize + complex match + w_int + 12x1536 gate matmul +
// sigmoid gate + RMSNorm -> g (B,L,1536).  One workgroup per token.
// ---------------------------------------------------------------------------
__global__ __launch_bounds__(256)
void state_gate_kernel(const float* __restrict__ cs, const float* __restrict__ qn,
                       const float* __restrict__ zall, const float* __restrict__ W_gate,
                       const float* __restrict__ gnorm_w, const float* __restrict__ w_int_raw,
                       float* __restrict__ g)
{
    __shared__ float oc[1536];
    __shared__ float invd[12];
    __shared__ float gl[12];
    __shared__ float sred;

    const int    tid = threadIdx.x;
    const size_t bl  = blockIdx.x;

    if (tid < 12) {
        invd[tid] = 1.0f / fmaxf(cs[bl * 1548 + tid], 1e-4f);
        gl[tid]   = zall[bl * 1752 + 1740 + tid];
    }
    if (tid == 0) sred = 0.0f;
    __syncthreads();

    for (int t = tid; t < 768; t += 256) {
        int   k  = t >> 6, h = t & 63, kq = k >> 1;
        float id = invd[k];
        float sre = cs[bl * 1548 + 12  + t] * id;
        float sim = cs[bl * 1548 + 780 + t] * id;
        float qre = qn[bl * 768 + kq * 128 + h * 2 + 0];
        float qim = qn[bl * 768 + kq * 128 + h * 2 + 1];
        float wi  = __expf(fminf(fmaxf(w_int_raw[t], -5.0f), 5.0f));
        float ore = (sre * qre + sim * qim) * wi;
        float oim = (sim * qre - sre * qim) * wi;
        int c1 = k * 128 + h, c2 = c1 + 64;
        float g1a = 0.0f, g2a = 0.0f;
        #pragma unroll
        for (int j = 0; j < 12; ++j) {
            g1a += gl[j] * W_gate[j * 1536 + c1];
            g2a += gl[j] * W_gate[j * 1536 + c2];
        }
        float g1 = ore * sigmoidf_(g1a);
        float g2 = oim * sigmoidf_(g2a);
        oc[c1] = g1;
        oc[c2] = g2;
        atomicAdd(&sred, g1 * g1 + g2 * g2);
    }
    __syncthreads();

    float rn = rsqrtf(sred * (1.0f / 1536.0f) + 1e-6f);
    for (int c = tid; c < 1536; c += 256)
        g[bl * 1536 + c] = oc[c] * rn * gnorm_w[c];
}

// ---------------------------------------------------------------------------
__global__ void hs_scale_kernel(float* __restrict__ y, const float* __restrict__ hs, long n)
{
    long i = (long)blockIdx.x * blockDim.x + threadIdx.x;
    if (i >= n) return;
    int k = (int)((i % 4608) / 384);
    y[i] *= hs[k];
}

__global__ void swiglu_kernel(const float* __restrict__ yc, float* __restrict__ y, long n)
{
    long i = (long)blockIdx.x * blockDim.x + threadIdx.x;
    if (i >= n) return;
    long bl = i / 2304;
    int  r  = (int)(i % 2304);
    int  k  = r / 192, j = r % 192;
    size_t base = (size_t)bl * 4608 + (size_t)k * 384;
    float v  = yc[base + j];
    float gt = yc[base + 192 + j];
    y[i] = v * gt * sigmoidf_(gt);
}

// ---------------------------------------------------------------------------
extern "C" void kernel_launch(void* const* d_in, const int* in_sizes, int n_in,
                              void* d_out, int out_size, void* d_ws, size_t ws_size,
                              hipStream_t stream)
{
    const float* x           = (const float*)d_in[0];
    const float* W_in        = (const float*)d_in[1];
    const float* conv_w      = (const float*)d_in[2];
    const float* k_norm_w    = (const float*)d_in[3];
    const float* q_norm_w    = (const float*)d_in[4];
    const float* theta_raw   = (const float*)d_in[5];
    const float* w_int_raw   = (const float*)d_in[6];
    const float* decay       = (const float*)d_in[7];
    const float* score_scale = (const float*)d_in[8];
    const float* score_bias  = (const float*)d_in[9];
    const float* tanh_scale  = (const float*)d_in[10];
    const float* W_readout   = (const float*)d_in[11];
    const float* W_gate      = (const float*)d_in[12];
    const float* gnorm_w     = (const float*)d_in[13];
    const float* W_skip      = (const float*)d_in[14];
    const float* hscale      = (const float*)d_in[15];
    const float* W_out       = (const float*)d_in[16];

    const int  L  = 4096;
    const long BL = (long)in_sizes[0] / 768;   // B*L = 16384
    const int  Bn = (int)(BL / L);

    // f32 scratch
    float* ws    = (float*)d_ws;
    float* zall  = ws;                 // BL*1752
    float* stack = zall  + BL * 1752;  // BL*1548 (cumsum in place)
    float* qn    = stack + BL * 1548;  // BL*768
    float* g     = qn    + BL * 768;   // BL*1536
    float* ycomb = g     + BL * 1536;  // BL*4608
    float* ybuf  = ycomb + BL * 4608;  // BL*2304
    // bf16 scratch (GEMM operands, K-contiguous)
    __bf16* x_bf   = (__bf16*)(ybuf + BL * 2304);   // BL*768
    __bf16* cs_bf  = x_bf  + BL * 768;              // BL*192
    __bf16* g_bf   = cs_bf + BL * 192;              // BL*1536
    __bf16* y_bf   = g_bf  + BL * 1536;             // BL*2304
    __bf16* WinT   = y_bf  + BL * 2304;             // 1752*768
    __bf16* WskipT = WinT   + 1752L * 768;          // 4608*192
    __bf16* WroT   = WskipT + 4608L * 192;          // 12*384*128
    __bf16* WoutT  = WroT   + 12L * 384 * 128;      // 768*2304

    const dim3 gblk(128);
    const int  Mt = (int)((BL + 63) / 64);
    auto blocks1d = [](long n) { return dim3((unsigned)((n + 255) / 256)); };

    // 0) one-shot conversions / weight transposes to bf16
    cvt_bf16_kernel<<<blocks1d(BL * 768), dim3(256), 0, stream>>>(x, x_bf, BL * 768);
    transpose_cvt_kernel<<<dim3((unsigned)((768L * 1752 + 255) / 256), 1), dim3(256), 0, stream>>>(
        W_in, WinT, 768, 1752, 0, 0);
    transpose_cvt_kernel<<<dim3((unsigned)((192L * 4608 + 255) / 256), 1), dim3(256), 0, stream>>>(
        W_skip, WskipT, 192, 4608, 0, 0);
    transpose_cvt_kernel<<<dim3((unsigned)((128L * 384 + 255) / 256), 12), dim3(256), 0, stream>>>(
        W_readout, WroT, 128, 384, 49152, 49152);
    transpose_cvt_kernel<<<dim3((unsigned)((2304L * 768 + 255) / 256), 1), dim3(256), 0, stream>>>(
        W_out, WoutT, 2304, 768, 0, 0);

    // 1) z_all = x @ W_in                        (16384x1752, K=768)
    gemm_bf16_wmma_tdm<<<dim3((1752 + 63) / 64, Mt, 1), gblk, 0, stream>>>(
        x_bf, WinT, zall, (int)BL, 1752, 768, 768, 768, 1752, 0, 0, 0, 0);

    // 2) conv + SiLU + norms + stack production
    conv_stack_kernel<<<dim3((unsigned)BL), dim3(256), 0, stream>>>(
        zall, conv_w, k_norm_w, q_norm_w, theta_raw, decay,
        score_scale, score_bias, tanh_scale, stack, qn, L);

    // 3) cumsum along L (in place)
    cumsum_kernel<<<dim3(7, Bn), dim3(256), 0, stream>>>(stack, L);

    // 4) state + gate + gnorm -> g ; compact c_skip -> bf16
    state_gate_kernel<<<dim3((unsigned)BL), dim3(256), 0, stream>>>(
        stack, qn, zall, W_gate, gnorm_w, w_int_raw, g);
    gather_cskip_kernel<<<blocks1d(BL * 192), dim3(256), 0, stream>>>(zall, cs_bf, BL * 192);
    cvt_bf16_kernel<<<blocks1d(BL * 1536), dim3(256), 0, stream>>>(g, g_bf, BL * 1536);

    // 5) y_skip = c_skip @ W_skip               (16384x4608, K=192)
    gemm_bf16_wmma_tdm<<<dim3(4608 / 64, Mt, 1), gblk, 0, stream>>>(
        cs_bf, WskipT, ycomb, (int)BL, 4608, 192, 192, 192, 4608, 0, 0, 0, 0);

    // 6) ycomb *= highway_scale[k]
    hs_scale_kernel<<<blocks1d(BL * 4608), dim3(256), 0, stream>>>(ycomb, hscale, BL * 4608);

    // 7) ycomb += g_k @ W_readout_k  (batched over k=12; 16384x384, K=128)
    gemm_bf16_wmma_tdm<<<dim3(384 / 64, Mt, 12), gblk, 0, stream>>>(
        g_bf, WroT, ycomb, (int)BL, 384, 128, 1536, 128, 4608,
        128L, 49152L, 384L, 1);

    // 8) SwiGLU combine -> y (BL x 2304), then bf16
    swiglu_kernel<<<blocks1d(BL * 2304), dim3(256), 0, stream>>>(ycomb, ybuf, BL * 2304);
    cvt_bf16_kernel<<<blocks1d(BL * 2304), dim3(256), 0, stream>>>(ybuf, y_bf, BL * 2304);

    // 9) out = y @ W_out                         (16384x768, K=2304)
    gemm_bf16_wmma_tdm<<<dim3(768 / 64, Mt, 1), gblk, 0, stream>>>(
        y_bf, WoutT, (float*)d_out, (int)BL, 768, 2304, 2304, 2304, 768, 0, 0, 0, 0);
}
